// GAPModel_40742059770459
// MI455X (gfx1250) — matrix operand order repaired and checked by reading, not compiled, not run on hardware
//
#include <hip/hip_runtime.h>
#include <hip/hip_bf16.h>
#include <math.h>

#define N_NODES 50000
#define N_EDGES 800000
#define IN_FEATS 128
#define SAGE_HID 512
#define H_FEATS 128
#define MLP_HID 32
#define N_PARTS 8

typedef __attribute__((ext_vector_type(16))) __bf16 v16bf;
typedef __attribute__((ext_vector_type(8)))  __bf16 v8bf;
typedef __attribute__((ext_vector_type(8)))  float  v8f;

// ---------------- utility kernels ----------------

__global__ __launch_bounds__(256) void k_zero(float* p, long long n) {
  long long i = (long long)blockIdx.x * blockDim.x + threadIdx.x;
  long long stride = (long long)gridDim.x * blockDim.x;
  for (; i < n; i += stride) p[i] = 0.0f;
}

__global__ __launch_bounds__(256) void k_degree(const int* __restrict__ dst,
                                                float* __restrict__ deg) {
  int e = blockIdx.x * blockDim.x + threadIdx.x;
  if (e < N_EDGES) atomicAdd(&deg[dst[e]], 1.0f);
}

__global__ __launch_bounds__(256) void k_deg_inv(float* deg) {
  int i = blockIdx.x * blockDim.x + threadIdx.x;
  if (i < N_NODES) { float d = deg[i]; deg[i] = 1.0f / (d > 1.0f ? d : 1.0f); }
}

// One wave (32 lanes) per edge: each lane moves a float4 (128 floats total).
__global__ __launch_bounds__(256) void k_edge_agg128(const int* __restrict__ src,
                                                     const int* __restrict__ dst,
                                                     const float* __restrict__ x,
                                                     int ldx, int offx,
                                                     float* __restrict__ agg) {
  int wave = threadIdx.x >> 5, lane = threadIdx.x & 31;
  int e = blockIdx.x * 8 + wave;
  if (e >= N_EDGES) return;
  int s = src[e], d = dst[e];
  const float4 v = *(const float4*)(x + (size_t)s * ldx + offx + lane * 4);
  float* ap = agg + (size_t)d * 128 + lane * 4;
  atomicAdd(ap + 0, v.x);
  atomicAdd(ap + 1, v.y);
  atomicAdd(ap + 2, v.z);
  atomicAdd(ap + 3, v.w);
}

// A1[n, 0:128] = feats[n]; A1[n, 128:256] = agg1[n] * deg_inv[n]   (bf16)
__global__ __launch_bounds__(256) void k_prep_A1(const float* __restrict__ feats,
                                                 const float* __restrict__ agg1,
                                                 const float* __restrict__ deginv,
                                                 __bf16* __restrict__ A1) {
  int i = blockIdx.x * blockDim.x + threadIdx.x;
  if (i >= N_NODES * IN_FEATS) return;
  int n = i >> 7, f = i & 127;
  A1[(size_t)n * 256 + f]       = (__bf16)feats[i];
  A1[(size_t)n * 256 + 128 + f] = (__bf16)(agg1[i] * deginv[n]);
}

// W1cat (256x512): rows 0..127 = W1_self, rows 128..255 = W1_neigh
__global__ __launch_bounds__(256) void k_prep_W1(const float* __restrict__ Ws,
                                                 const float* __restrict__ Wn,
                                                 __bf16* __restrict__ W) {
  int i = blockIdx.x * blockDim.x + threadIdx.x;
  if (i >= 256 * 512) return;
  int k = i >> 9, n = i & 511;
  W[i] = (__bf16)(k < 128 ? Ws[k * 512 + n] : Wn[(k - 128) * 512 + n]);
}

// W2cat (512x256): cols 0..127 = W2_self, cols 128..255 = W2_neigh
__global__ __launch_bounds__(256) void k_prep_W2(const float* __restrict__ Ws,
                                                 const float* __restrict__ Wn,
                                                 __bf16* __restrict__ W) {
  int i = blockIdx.x * blockDim.x + threadIdx.x;
  if (i >= 512 * 256) return;
  int k = i >> 8, n = i & 255;
  W[i] = (__bf16)(n < 128 ? Ws[k * 128 + n] : Wn[k * 128 + (n - 128)]);
}

// ---------------- WMMA bf16 GEMM (16x64 tile per wave, 4 accumulators) -----
// Fragment layouts per CDNA5 ISA 7.12.2:
//   A (16x32 bf16): lane l: M = l%16, g = l/16; elem e<8 -> K = kb + g*8 + e,
//                   e>=8 -> K = kb + 16 + g*8 + (e-8)   (two 8-elem runs)
//   B (32x16 bf16): lane l: N = l%16, g = l/16; elem e -> same K as A elem e
//   C (16x16 f32):  elem c: row = c + 8*(l/16), col = l%16
// A fragment is loaded once per K-step and reused across 4 adjacent B tiles:
// load:wmma ratio 2.5:1 instead of 4:1.
template <bool BIAS_RELU, typename OutT>
__global__ __launch_bounds__(256) void k_gemm4(const __bf16* __restrict__ A,
                                               const __bf16* __restrict__ W,
                                               const float* __restrict__ bias,
                                               OutT* __restrict__ out,
                                               int M, int N, int K) {
  int wave = threadIdx.x >> 5, lane = threadIdx.x & 31;
  int ngroups = N >> 6;  // 64-wide N groups
  int tile = blockIdx.x * 8 + wave;
  int tm = tile / ngroups;
  int tg = tile - tm * ngroups;
  if (tm * 16 >= M) return;  // wave-uniform; EXEC stays all-ones for WMMA

  int g = lane >> 4, mr = lane & 15;
  const __bf16* Arow  = A + (size_t)(tm * 16 + mr) * K;
  const __bf16* Wbase = W + tg * 64;
  v8f acc0 = {}, acc1 = {}, acc2 = {}, acc3 = {};
#pragma unroll 2
  for (int kb = 0; kb < K; kb += 32) {
    v8bf lo = *(const v8bf*)(Arow + kb + g * 8);
    v8bf hi = *(const v8bf*)(Arow + kb + 16 + g * 8);
    v16bf a;
#pragma unroll
    for (int i = 0; i < 8; ++i) { a[i] = lo[i]; a[8 + i] = hi[i]; }
    v16bf b0, b1, b2, b3;
#pragma unroll
    for (int i = 0; i < 16; ++i) {
      int kk = kb + (i < 8 ? g * 8 + i : 16 + g * 8 + (i - 8));
      const __bf16* wr = Wbase + (size_t)kk * N + mr;
      b0[i] = wr[0]; b1[i] = wr[16]; b2[i] = wr[32]; b3[i] = wr[48];
    }
    acc0 = __builtin_amdgcn_wmma_f32_16x16x32_bf16(false, a, false, b0, (short)0,
                                                   acc0, false, false);
    acc1 = __builtin_amdgcn_wmma_f32_16x16x32_bf16(false, a, false, b1, (short)0,
                                                   acc1, false, false);
    acc2 = __builtin_amdgcn_wmma_f32_16x16x32_bf16(false, a, false, b2, (short)0,
                                                   acc2, false, false);
    acc3 = __builtin_amdgcn_wmma_f32_16x16x32_bf16(false, a, false, b3, (short)0,
                                                   acc3, false, false);
  }
  v8f accs[4] = {acc0, acc1, acc2, acc3};
#pragma unroll
  for (int t = 0; t < 4; ++t) {
    int col = tg * 64 + t * 16 + mr;
    float bv = BIAS_RELU ? bias[col] : 0.0f;
#pragma unroll
    for (int c = 0; c < 8; ++c) {
      int row = tm * 16 + c + 8 * g;
      float v = accs[t][c] + bv;
      if (BIAS_RELU) v = v > 0.0f ? v : 0.0f;
      out[(size_t)row * N + col] = (OutT)v;
    }
  }
}

// ---------------- fused tail: relu(self+b2+agg2*dinv) -> MLP -> softmax ----
__global__ __launch_bounds__(256) void k_final(const float* __restrict__ T,
                                               const float* __restrict__ agg2,
                                               const float* __restrict__ deginv,
                                               const float* __restrict__ b2,
                                               const float* __restrict__ Wp1,
                                               const float* __restrict__ bp1,
                                               const float* __restrict__ Wp2,
                                               const float* __restrict__ bp2,
                                               float* __restrict__ out) {
  __shared__ float hsh[8][128];
  __shared__ float xsh[8][32];
  __shared__ float lsh[8][8];
  int wave = threadIdx.x >> 5, lane = threadIdx.x & 31;
  int node = blockIdx.x * 8 + wave;
  bool ok = node < N_NODES;
  if (ok) {
    float di = deginv[node];
#pragma unroll
    for (int k = 0; k < 4; ++k) {
      int f = lane * 4 + k;
      float sv = T[(size_t)node * 256 + f] + b2[f];           // self part
      float av = agg2[(size_t)node * 128 + f] * di;           // mean-neigh part
      float h = sv + av;
      hsh[wave][f] = h > 0.0f ? h : 0.0f;
    }
  }
  __syncthreads();
  if (ok) {
    float acc = bp1[lane];  // MLP_HID == 32 == lanes
#pragma unroll 8
    for (int f = 0; f < 128; ++f) acc += hsh[wave][f] * Wp1[f * 32 + lane];
    xsh[wave][lane] = acc > 0.0f ? acc : 0.0f;
  }
  __syncthreads();
  if (ok && lane < 8) {
    float lg = bp2[lane];
#pragma unroll
    for (int j = 0; j < 32; ++j) lg += xsh[wave][j] * Wp2[j * 8 + lane];
    lsh[wave][lane] = lg;
  }
  __syncthreads();
  if (ok && lane < 8) {
    float m = lsh[wave][0];
#pragma unroll
    for (int p = 1; p < 8; ++p) m = fmaxf(m, lsh[wave][p]);
    float s = 0.0f;
#pragma unroll
    for (int p = 0; p < 8; ++p) s += __expf(lsh[wave][p] - m);
    out[(size_t)node * 8 + lane] = __expf(lsh[wave][lane] - m) / s;
  }
}

// ---------------- launcher ----------------

extern "C" void kernel_launch(void* const* d_in, const int* in_sizes, int n_in,
                              void* d_out, int out_size, void* d_ws, size_t ws_size,
                              hipStream_t stream) {
  const float* feats = (const float*)d_in[0];
  const int*   src   = (const int*)d_in[1];
  const int*   dst   = (const int*)d_in[2];
  const float* W1s   = (const float*)d_in[3];
  const float* W1n   = (const float*)d_in[4];
  const float* b1    = (const float*)d_in[5];
  const float* W2s   = (const float*)d_in[6];
  const float* W2n   = (const float*)d_in[7];
  const float* b2    = (const float*)d_in[8];
  const float* Wp1   = (const float*)d_in[9];
  const float* bp1   = (const float*)d_in[10];
  const float* Wp2   = (const float*)d_in[11];
  const float* bp2   = (const float*)d_in[12];
  char* ws = (char*)d_ws;

  auto align = [](size_t x) { return (x + 255) & ~(size_t)255; };
  size_t o_deg  = 0;                        size_t sz_deg  = (size_t)N_NODES * 4;
  size_t o_agg1 = align(o_deg + sz_deg);    size_t sz_agg1 = (size_t)N_NODES * 128 * 4;
  size_t o_A1   = align(o_agg1 + sz_agg1);  size_t sz_A1   = (size_t)N_NODES * 256 * 2;
  size_t o_T    = o_agg1;                   // T (50000x256 f32) overlaps dead agg1+A1
  size_t o_h1   = align(o_A1 + sz_A1);      size_t sz_h1   = (size_t)N_NODES * 512 * 2;
  size_t o_agg2 = o_h1;                     // agg2 overlaps dead h1
  size_t o_W1   = align(o_h1 + sz_h1);      size_t sz_W1   = (size_t)256 * 512 * 2;
  size_t o_W2   = align(o_W1 + sz_W1);

  float*  deg  = (float*)(ws + o_deg);
  float*  agg1 = (float*)(ws + o_agg1);
  __bf16* A1   = (__bf16*)(ws + o_A1);
  float*  T    = (float*)(ws + o_T);
  __bf16* h1   = (__bf16*)(ws + o_h1);
  float*  agg2 = (float*)(ws + o_agg2);
  __bf16* W1c  = (__bf16*)(ws + o_W1);
  __bf16* W2c  = (__bf16*)(ws + o_W2);

  // 1) zero deg + agg1 (contiguous span)
  long long nz1 = (long long)((o_agg1 + sz_agg1) / 4);
  k_zero<<<2048, 256, 0, stream>>>((float*)ws, nz1);
  // 2) degree + layer-1 aggregation (f32 atomics into L2)
  k_degree<<<(N_EDGES + 255) / 256, 256, 0, stream>>>(dst, deg);
  k_edge_agg128<<<(N_EDGES + 7) / 8, 256, 0, stream>>>(src, dst, feats, 128, 0, agg1);
  k_deg_inv<<<(N_NODES + 255) / 256, 256, 0, stream>>>(deg);
  // 3) bf16 operand prep
  k_prep_W1<<<(256 * 512 + 255) / 256, 256, 0, stream>>>(W1s, W1n, W1c);
  k_prep_W2<<<(512 * 256 + 255) / 256, 256, 0, stream>>>(W2s, W2n, W2c);
  k_prep_A1<<<(N_NODES * 128 + 255) / 256, 256, 0, stream>>>(feats, agg1, deg, A1);
  // 4) h1 = relu([x|agg] @ [W1_self;W1_neigh] + b1)  -> bf16  (M=50000,N=512,K=256)
  //    tiles = 3125 * (512/64) = 25000 waves -> 3125 blocks
  k_gemm4<true, __bf16><<<3125, 256, 0, stream>>>(A1, W1c, b1, h1, N_NODES, 512, 256);
  // 5) T = h1 @ [W2_self|W2_neigh]  -> f32  (M=50000,N=256,K=512)
  //    tiles = 3125 * (256/64) = 12500 waves -> 1563 blocks (tail guarded)
  k_gemm4<false, float><<<(12500 + 7) / 8, 256, 0, stream>>>(h1, W2c, nullptr, T,
                                                             N_NODES, 256, 512);
  // 6) layer-2 aggregation on the 128-dim neigh projection
  k_zero<<<2048, 256, 0, stream>>>(agg2, (long long)N_NODES * 128);
  k_edge_agg128<<<(N_EDGES + 7) / 8, 256, 0, stream>>>(src, dst, T, 256, 128, agg2);
  // 7) fused relu + MLP + softmax
  k_final<<<(N_NODES + 7) / 8, 256, 0, stream>>>(T, agg2, deg, b2, Wp1, bp1, Wp2,
                                                 bp2, (float*)d_out);
}
